// MultiHeadAttention_25778393711041
// MI455X (gfx1250) — compile-verified, matrix operand order checked by reading, not run on hardware
//
#include <hip/hip_runtime.h>

typedef __attribute__((ext_vector_type(16))) __bf16 v16bf;
typedef __attribute__((ext_vector_type(8)))  __bf16 v8bf;
typedef __attribute__((ext_vector_type(8)))  float  v8f;
typedef int v4i_vs __attribute__((vector_size(16)));  // matches builtin proto

#define DI __device__ __forceinline__

#if __has_builtin(__builtin_amdgcn_global_load_async_to_lds_b128)
#define HAVE_ASYNC_CP 1
#else
#define HAVE_ASYNC_CP 0
#endif

// 16-byte global -> LDS copy. On CDNA5 use the async tensor path
// (ASYNCcnt-tracked, no VGPR bounce, overlaps with WMMA); otherwise
// fall back to a synchronous copy through registers.
DI void cp_async16(__bf16* lds, const __bf16* g) {
#if HAVE_ASYNC_CP
  __builtin_amdgcn_global_load_async_to_lds_b128(
      (__attribute__((address_space(1))) v4i_vs*)g,
      (__attribute__((address_space(3))) v4i_vs*)lds, 0, 0);
#else
  *(uint4*)lds = *(const uint4*)g;
#endif
}

template <int N>
DI void wait_async() {
#if HAVE_ASYNC_CP
#if __has_builtin(__builtin_amdgcn_s_wait_asynccnt)
  __builtin_amdgcn_s_wait_asynccnt((unsigned short)N);
#else
  asm volatile("s_wait_asynccnt %0" ::"i"(N) : "memory");
#endif
#endif
}

// D = A(16x32 bf16) x B(32x16 bf16) + C(16x16 f32)
DI v8f wmma_bf16(v16bf a, v16bf b, v8f c) {
  return __builtin_amdgcn_wmma_f32_16x16x32_bf16(false, a, false, b, (short)0, c,
                                                 false, false);
}

// Fragment for 16x16x32 WMMA: lane (idx16 + 16*half) holds row elements
// k = {8*half+0..7, 8*half+16..23} of a 32-wide k-chunk -> two 16B loads.
DI v16bf frag_ld(const __bf16* p) {
  v8bf x = *(const v8bf*)p;
  v8bf y = *(const v8bf*)(p + 16);
  v16bf r;
#pragma unroll
  for (int i = 0; i < 8; ++i) { r[i] = x[i]; r[i + 8] = y[i]; }
  return r;
}

DI v8f zero8() {
  v8f z;
#pragma unroll
  for (int i = 0; i < 8; ++i) z[i] = 0.f;
  return z;
}

DI void split2(float v, __bf16& h, __bf16& l) {
  h = (__bf16)v;
  l = (__bf16)(v - (float)h);
}

// ---------------------------------------------------------------------------
// GEMM mainloop: C[128x128] tile, K-step 32, 8 waves (2m x 4n), each wave
// a 64x32 sub-tile (4x2 accum frags). A and B are pre-split bf16 hi/lo; A in
// [M][K], B transposed [N][K] (fragment-friendly). Async double-buffered LDS
// staging; split-bf16 fp32 emulation: acc += Ah*Bh + Al*Bh + Ah*Bl.
// LDS row stride 40 bf16 (80B): 16B-aligned rows, conflict-free b128 reads.
// ---------------------------------------------------------------------------
DI void stage_gemm(const __bf16* __restrict__ Ah, const __bf16* __restrict__ Al,
                   int lda, int m0, const __bf16* __restrict__ Bh,
                   const __bf16* __restrict__ Bl, int ldb, int n0, int kt,
                   __bf16* dAh, __bf16* dAl, __bf16* dBh, __bf16* dBl,
                   int tid) {
  const int slot = tid & 3;  // 4 x 16B per 64B row
  const int r0 = tid >> 2;   // 64 rows per pass
#pragma unroll
  for (int i = 0; i < 2; ++i) {  // 8 async instructions per wave per tile
    const int row = r0 + i * 64;
    const size_t ga = (size_t)(m0 + row) * lda + kt + slot * 8;
    const size_t gb = (size_t)(n0 + row) * ldb + kt + slot * 8;
    const int ld = row * 40 + slot * 8;
    cp_async16(dAh + ld, Ah + ga);
    cp_async16(dAl + ld, Al + ga);
    cp_async16(dBh + ld, Bh + gb);
    cp_async16(dBl + ld, Bl + gb);
  }
}

DI void gemm_bf16x3_mainloop(const __bf16* __restrict__ Ah,
                             const __bf16* __restrict__ Al, int lda,
                             const __bf16* __restrict__ Bh,
                             const __bf16* __restrict__ Bl, int ldb, int K,
                             int m0, int n0, __bf16* sAh, __bf16* sAl,
                             __bf16* sBh, __bf16* sBl, v8f (&acc)[4][2],
                             int tid) {
  const int lane = tid & 31, wave = tid >> 5;
  const int wm = wave & 1, wn = wave >> 1;
  const int n16 = lane & 15, lh = lane >> 4;
  const int ntiles = K >> 5;
  constexpr int BUF = 128 * 40;

  stage_gemm(Ah, Al, lda, m0, Bh, Bl, ldb, n0, 0, sAh, sAl, sBh, sBl, tid);

  for (int kt = 0; kt < ntiles; ++kt) {
    const int cur = kt & 1, nxt = cur ^ 1;
    if (kt + 1 < ntiles) {  // prefetch next tile, then drain current tile
      stage_gemm(Ah, Al, lda, m0, Bh, Bl, ldb, n0, (kt + 1) << 5,
                 sAh + nxt * BUF, sAl + nxt * BUF, sBh + nxt * BUF,
                 sBl + nxt * BUF, tid);
      wait_async<8>();
    } else {
      wait_async<0>();
    }
    __syncthreads();

    const __bf16* cAh = sAh + cur * BUF;
    const __bf16* cAl = sAl + cur * BUF;
    const __bf16* cBh = sBh + cur * BUF;
    const __bf16* cBl = sBl + cur * BUF;

    v16bf bh[2], bl[2];
#pragma unroll
    for (int nf = 0; nf < 2; ++nf) {
      const int off = (wn * 32 + nf * 16 + n16) * 40 + 8 * lh;
      bh[nf] = frag_ld(cBh + off);
      bl[nf] = frag_ld(cBl + off);
    }
#pragma unroll
    for (int mf = 0; mf < 4; ++mf) {
      const int off = (wm * 64 + mf * 16 + n16) * 40 + 8 * lh;
      v16bf ah = frag_ld(cAh + off);
      v16bf al = frag_ld(cAl + off);
#pragma unroll
      for (int nf = 0; nf < 2; ++nf) {
        acc[mf][nf] = wmma_bf16(ah, bh[nf], acc[mf][nf]);
        acc[mf][nf] = wmma_bf16(al, bh[nf], acc[mf][nf]);
        acc[mf][nf] = wmma_bf16(ah, bl[nf], acc[mf][nf]);
      }
    }
    __syncthreads();  // protects buffer reuse by next prefetch
  }
}

// ---------------------------------------------------------------------------
// Kernel 0a: hi/lo split (no transpose): X[idx] -> Xh/Xl
// ---------------------------------------------------------------------------
__global__ void split_kernel(const float* __restrict__ W,
                             __bf16* __restrict__ Th, __bf16* __restrict__ Tl,
                             unsigned n) {
  const unsigned idx = blockIdx.x * 256 + threadIdx.x;
  if (idx >= n) return;
  __bf16 h, l;
  split2(W[idx], h, l);
  Th[idx] = h;
  Tl[idx] = l;
}

// ---------------------------------------------------------------------------
// Kernel 0b: transpose + hi/lo split of a weight matrix W[K][N] -> T[N][K]
// ---------------------------------------------------------------------------
__global__ void split_transpose_kernel(const float* __restrict__ W,
                                       __bf16* __restrict__ Th,
                                       __bf16* __restrict__ Tl, int K, int N) {
  const size_t idx = (size_t)blockIdx.x * 256 + threadIdx.x;
  if (idx >= (size_t)K * N) return;
  const int k = (int)(idx / N), n = (int)(idx % N);
  __bf16 h, l;
  split2(W[idx], h, l);
  Th[(size_t)n * K + k] = h;
  Tl[(size_t)n * K + k] = l;
}

// ---------------------------------------------------------------------------
// Kernel 1: qkv = x @ W_attn + b_attn, fused RoPE; scatter to head-major
// Q/K [B,H,T,64] and transposed V [B,H,64,T], all bf16 hi/lo.
// ---------------------------------------------------------------------------
__global__ __launch_bounds__(256) void qkv_rope_kernel(
    const __bf16* __restrict__ Xh, const __bf16* __restrict__ Xl,
    const __bf16* __restrict__ WAth, const __bf16* __restrict__ WAtl,
    const float* __restrict__ b_attn, const float* __restrict__ rope,
    __bf16* __restrict__ Qh, __bf16* __restrict__ Ql,
    __bf16* __restrict__ Kh, __bf16* __restrict__ Kl,
    __bf16* __restrict__ Vth, __bf16* __restrict__ Vtl) {
  __shared__ __bf16 sAh[2 * 128 * 40], sAl[2 * 128 * 40];
  __shared__ __bf16 sBh[2 * 128 * 40], sBl[2 * 128 * 40];
  const int tid = threadIdx.x;
  const int m0 = blockIdx.x * 128, n0 = blockIdx.y * 128;
  v8f acc[4][2];
#pragma unroll
  for (int i = 0; i < 4; ++i)
#pragma unroll
    for (int j = 0; j < 2; ++j) acc[i][j] = zero8();

  gemm_bf16x3_mainloop(Xh, Xl, 1024, WAth, WAtl, 1024, 1024, m0, n0, sAh, sAl,
                       sBh, sBl, acc, tid);

  const int lane = tid & 31, wave = tid >> 5;
  const int wm = wave & 1, wn = wave >> 1;
  const int n16 = lane & 15, lh = lane >> 4;
#pragma unroll
  for (int mf = 0; mf < 4; ++mf)
#pragma unroll
    for (int nf = 0; nf < 2; ++nf) {
      const int gn = n0 + wn * 32 + nf * 16 + n16;  // region uniform per frag
      const float bias = b_attn[gn];
#pragma unroll
      for (int r = 0; r < 8; ++r) {
        const int gm = m0 + wm * 64 + mf * 16 + r + 8 * lh;  // 0..4095
        const int t = gm & 2047, bidx = gm >> 11;
        float val = acc[mf][nf][r] + bias;
        const float partner = __shfl_xor(val, 1, 32);  // RoPE pair (d^1)
        if (gn < 2048) {                               // q or k: apply RoPE
          const int nr = (gn < 1024) ? gn : gn - 1024;
          const int hh = nr >> 6, d = nr & 63, ii = d >> 1;
          const float cv = rope[((t << 5) + ii) * 4 + 0];  // [t,i,0,0]
          const float sv = rope[((t << 5) + ii) * 4 + 2];  // [t,i,1,0]
          const float out =
              (d & 1) ? (partner * sv + val * cv) : (val * cv - partner * sv);
          __bf16 oh, ol;
          split2(out, oh, ol);
          const size_t qi = (((size_t)(bidx * 16 + hh) * 2048 + t) << 6) + d;
          if (gn < 1024) { Qh[qi] = oh; Ql[qi] = ol; }
          else           { Kh[qi] = oh; Kl[qi] = ol; }
        } else {  // v: store transposed [b,h,d,t]
          const int nr = gn - 2048;
          const int hh = nr >> 6, d = nr & 63;
          __bf16 oh, ol;
          split2(val, oh, ol);
          const size_t vi = ((size_t)(bidx * 16 + hh) * 64 + d) * 2048 + t;
          Vth[vi] = oh;
          Vtl[vi] = ol;
        }
      }
    }
}

// ---------------------------------------------------------------------------
// Kernel 2: causal flash attention. Block = 8 waves x 32 q-rows = 256 rows.
// Async double-buffered KV tiles of 32; S and PV via split-bf16 WMMA; fp32
// online softmax with 16-lane butterfly row reductions. Output y emitted
// directly as bf16 hi/lo for the projection GEMM.
// ---------------------------------------------------------------------------
DI void stage_attn(const __bf16* __restrict__ gKh,
                   const __bf16* __restrict__ gKl,
                   const __bf16* __restrict__ gVh,
                   const __bf16* __restrict__ gVl, int k0, __bf16* dKh,
                   __bf16* dKl, __bf16* dVh, __bf16* dVl, int tid) {
  {  // K tile: 32 rows x 128B
    const int row = tid >> 3, slot = tid & 7;
    const size_t g = (size_t)(k0 + row) * 64 + slot * 8;
    const int l = row * 72 + slot * 8;
    cp_async16(dKh + l, gKh + g);
    cp_async16(dKl + l, gKl + g);
  }
  {  // V^T tile: 64 rows x 64B
    const int row = tid >> 2, slot = tid & 3;
    const size_t g = (size_t)row * 2048 + k0 + slot * 8;
    const int l = row * 40 + slot * 8;
    cp_async16(dVh + l, gVh + g);
    cp_async16(dVl + l, gVl + g);
  }  // 4 async instructions per wave per tile
}

__global__ __launch_bounds__(256) void attn_kernel(
    const __bf16* __restrict__ Qh, const __bf16* __restrict__ Ql,
    const __bf16* __restrict__ Kh, const __bf16* __restrict__ Kl,
    const __bf16* __restrict__ Vth, const __bf16* __restrict__ Vtl,
    __bf16* __restrict__ Yh, __bf16* __restrict__ Yl) {
  __shared__ __bf16 sKh[2 * 32 * 72], sKl[2 * 32 * 72];  // [tk][d], stride 72
  __shared__ __bf16 sVh[2 * 64 * 40], sVl[2 * 64 * 40];  // [d][tk], stride 40
  __shared__ __bf16 sPh[8][32 * 40], sPl[8][32 * 40];    // per-wave P scratch
  constexpr int KBUF = 32 * 72, VBUF = 64 * 40;

  const int tid = threadIdx.x;
  const int lane = tid & 31, wave = tid >> 5;
  const int n16 = lane & 15, lh = lane >> 4;
  const int bh = blockIdx.y;
  const int q0 = blockIdx.x * 256;
  const int qw = q0 + wave * 32;

  const size_t base = (size_t)bh * 2048 * 64;
  const __bf16* gQh = Qh + base;
  const __bf16* gQl = Ql + base;
  const __bf16* gKh = Kh + base;
  const __bf16* gKl = Kl + base;
  const __bf16* gVh = Vth + base;  // [64][2048]
  const __bf16* gVl = Vtl + base;

  // Persistent Q fragments: [mf 0..1][k-chunk 0..1], hi/lo
  v16bf qfh[2][2], qfl[2][2];
#pragma unroll
  for (int mf = 0; mf < 2; ++mf)
#pragma unroll
    for (int kc = 0; kc < 2; ++kc) {
      const size_t off = (size_t)(qw + mf * 16 + n16) * 64 + kc * 32 + 8 * lh;
      qfh[mf][kc] = frag_ld(gQh + off);
      qfl[mf][kc] = frag_ld(gQl + off);
    }

  v8f o[2][4];
  float mst[2][8], lst[2][8];
#pragma unroll
  for (int mf = 0; mf < 2; ++mf) {
#pragma unroll
    for (int nd = 0; nd < 4; ++nd) o[mf][nd] = zero8();
#pragma unroll
    for (int r = 0; r < 8; ++r) {
      mst[mf][r] = -__builtin_inff();
      lst[mf][r] = 0.f;
    }
  }

  const int nkt = (q0 + 256) >> 5;  // causal: only kv tiles touching block
  stage_attn(gKh, gKl, gVh, gVl, 0, sKh, sKl, sVh, sVl, tid);

  for (int kt = 0; kt < nkt; ++kt) {
    const int k0 = kt << 5;
    const int cur = kt & 1, nxt = cur ^ 1;
    if (kt + 1 < nkt) {
      stage_attn(gKh, gKl, gVh, gVl, (kt + 1) << 5, sKh + nxt * KBUF,
                 sKl + nxt * KBUF, sVh + nxt * VBUF, sVl + nxt * VBUF, tid);
      wait_async<4>();
    } else {
      wait_async<0>();
    }
    __syncthreads();

    if (k0 <= qw + 31) {  // wave-uniform: skip fully-masked kv tiles
      const __bf16* cKh = sKh + cur * KBUF;
      const __bf16* cKl = sKl + cur * KBUF;
      const __bf16* cVh = sVh + cur * VBUF;
      const __bf16* cVl = sVl + cur * VBUF;

      v16bf kfh[2][2], kfl[2][2];
#pragma unroll
      for (int nf = 0; nf < 2; ++nf)
#pragma unroll
        for (int kc = 0; kc < 2; ++kc) {
          const int off = (nf * 16 + n16) * 72 + kc * 32 + 8 * lh;
          kfh[nf][kc] = frag_ld(cKh + off);
          kfl[nf][kc] = frag_ld(cKl + off);
        }
      // S = Q K^T (split-bf16)
      v8f s[2][2];
#pragma unroll
      for (int mf = 0; mf < 2; ++mf)
#pragma unroll
        for (int nf = 0; nf < 2; ++nf) {
          v8f a = zero8();
#pragma unroll
          for (int kc = 0; kc < 2; ++kc) {
            a = wmma_bf16(qfh[mf][kc], kfh[nf][kc], a);
            a = wmma_bf16(qfl[mf][kc], kfh[nf][kc], a);
            a = wmma_bf16(qfh[mf][kc], kfl[nf][kc], a);
          }
          s[mf][nf] = a;
        }
      // scale + causal mask + online softmax
#pragma unroll
      for (int mf = 0; mf < 2; ++mf) {
        float rmax[8];
#pragma unroll
        for (int r = 0; r < 8; ++r) {
          const int tq = qw + mf * 16 + r + 8 * lh;
          float mx = -__builtin_inff();
#pragma unroll
          for (int nf = 0; nf < 2; ++nf) {
            const int tk = k0 + nf * 16 + n16;
            float v = s[mf][nf][r] * 0.125f;  // 1/sqrt(64)
            if (tk > tq) v = -__builtin_inff();
            s[mf][nf][r] = v;
            mx = fmaxf(mx, v);
          }
          rmax[r] = mx;
        }
#pragma unroll
        for (int r = 0; r < 8; ++r)
#pragma unroll
          for (int off = 1; off < 16; off <<= 1)
            rmax[r] = fmaxf(rmax[r], __shfl_xor(rmax[r], off, 32));
#pragma unroll
        for (int r = 0; r < 8; ++r) {
          const float mnew = fmaxf(mst[mf][r], rmax[r]);
          const float corr = __expf(mst[mf][r] - mnew);
          float rsum = 0.f;
#pragma unroll
          for (int nf = 0; nf < 2; ++nf) {
            const float p = __expf(s[mf][nf][r] - mnew);
            s[mf][nf][r] = p;
            rsum += p;
          }
#pragma unroll
          for (int off = 1; off < 16; off <<= 1)
            rsum += __shfl_xor(rsum, off, 32);
          lst[mf][r] = lst[mf][r] * corr + rsum;
          mst[mf][r] = mnew;
#pragma unroll
          for (int nd = 0; nd < 4; ++nd) o[mf][nd][r] *= corr;
        }
      }
      // P (C-layout) -> per-wave LDS (hi/lo) -> A-fragments
      __bf16* pph = sPh[wave];
      __bf16* ppl = sPl[wave];
#pragma unroll
      for (int mf = 0; mf < 2; ++mf)
#pragma unroll
        for (int nf = 0; nf < 2; ++nf)
#pragma unroll
          for (int r = 0; r < 8; ++r) {
            const int row = mf * 16 + r + 8 * lh;
            const int col = nf * 16 + n16;
            __bf16 ph, pl;
            split2(s[mf][nf][r], ph, pl);
            pph[row * 40 + col] = ph;
            ppl[row * 40 + col] = pl;
          }
      // LDS is in-order within a wave: loads below see the stores above.
      v16bf pfh[2], pfl[2];
#pragma unroll
      for (int mf = 0; mf < 2; ++mf) {
        const int off = (mf * 16 + n16) * 40 + 8 * lh;
        pfh[mf] = frag_ld(pph + off);
        pfl[mf] = frag_ld(ppl + off);
      }
      v16bf vfh[4], vfl[4];
#pragma unroll
      for (int nd = 0; nd < 4; ++nd) {
        const int off = (nd * 16 + n16) * 40 + 8 * lh;
        vfh[nd] = frag_ld(cVh + off);
        vfl[nd] = frag_ld(cVl + off);
      }
#pragma unroll
      for (int mf = 0; mf < 2; ++mf)
#pragma unroll
        for (int nd = 0; nd < 4; ++nd) {
          o[mf][nd] = wmma_bf16(pfh[mf], vfh[nd], o[mf][nd]);
          o[mf][nd] = wmma_bf16(pfl[mf], vfh[nd], o[mf][nd]);
          o[mf][nd] = wmma_bf16(pfh[mf], vfl[nd], o[mf][nd]);
        }
    }
    __syncthreads();
  }

  // epilogue: normalize, hi/lo split, scatter to Yh/Yl [B,T,C] (bf16)
  const int bb = bh >> 4, hh = bh & 15;
#pragma unroll
  for (int mf = 0; mf < 2; ++mf)
#pragma unroll
    for (int r = 0; r < 8; ++r) {
      const float inv = 1.f / lst[mf][r];
      const int t = qw + mf * 16 + r + 8 * lh;
#pragma unroll
      for (int nd = 0; nd < 4; ++nd) {
        const int d = nd * 16 + n16;
        __bf16 oh, ol;
        split2(o[mf][nd][r] * inv, oh, ol);
        const size_t yi = ((size_t)bb * 2048 + t) * 1024 + hh * 64 + d;
        Yh[yi] = oh;
        Yl[yi] = ol;
      }
    }
}

// ---------------------------------------------------------------------------
// Kernel 3: out = Y @ W_proj + b_proj (fp32 out)
// ---------------------------------------------------------------------------
__global__ __launch_bounds__(256) void proj_kernel(
    const __bf16* __restrict__ Yh, const __bf16* __restrict__ Yl,
    const __bf16* __restrict__ WPh, const __bf16* __restrict__ WPl,
    const float* __restrict__ b_proj, float* __restrict__ out) {
  __shared__ __bf16 sAh[2 * 128 * 40], sAl[2 * 128 * 40];
  __shared__ __bf16 sBh[2 * 128 * 40], sBl[2 * 128 * 40];
  const int tid = threadIdx.x;
  const int m0 = blockIdx.x * 128, n0 = blockIdx.y * 128;
  v8f acc[4][2];
#pragma unroll
  for (int i = 0; i < 4; ++i)
#pragma unroll
    for (int j = 0; j < 2; ++j) acc[i][j] = zero8();

  gemm_bf16x3_mainloop(Yh, Yl, 1024, WPh, WPl, 1024, 1024, m0, n0, sAh, sAl,
                       sBh, sBl, acc, tid);

  const int lane = tid & 31, wave = tid >> 5;
  const int wm = wave & 1, wn = wave >> 1;
  const int n16 = lane & 15, lh = lane >> 4;
#pragma unroll
  for (int mf = 0; mf < 4; ++mf)
#pragma unroll
    for (int nf = 0; nf < 2; ++nf) {
      const int gn = n0 + wn * 32 + nf * 16 + n16;
      const float bias = b_proj[gn];
#pragma unroll
      for (int r = 0; r < 8; ++r) {
        const int gm = m0 + wm * 64 + mf * 16 + r + 8 * lh;
        out[(size_t)gm * 1024 + gn] = acc[mf][nf][r] + bias;
      }
    }
}

// ---------------------------------------------------------------------------
extern "C" void kernel_launch(void* const* d_in, const int* in_sizes, int n_in,
                              void* d_out, int out_size, void* d_ws,
                              size_t ws_size, hipStream_t stream) {
  (void)in_sizes; (void)n_in; (void)out_size; (void)ws_size;
  const float* x      = (const float*)d_in[0];
  const float* W_attn = (const float*)d_in[1];
  const float* b_attn = (const float*)d_in[2];
  const float* W_proj = (const float*)d_in[3];
  const float* b_proj = (const float*)d_in[4];
  const float* rope   = (const float*)d_in[5];

  char* ws = (char*)d_ws;
  size_t off = 0;
  auto alloc = [&](size_t bytes) -> void* {
    void* p = ws + off;
    off += (bytes + 255) & ~(size_t)255;
    return p;
  };
  const size_t QKV_ELEMS = (size_t)2 * 16 * 2048 * 64;  // B*H*T*HD
  const size_t XY_ELEMS  = (size_t)4096 * 1024;         // (B*T)*C
  __bf16* WAth = (__bf16*)alloc((size_t)3072 * 1024 * 2);
  __bf16* WAtl = (__bf16*)alloc((size_t)3072 * 1024 * 2);
  __bf16* WPth = (__bf16*)alloc((size_t)1024 * 1024 * 2);
  __bf16* WPtl = (__bf16*)alloc((size_t)1024 * 1024 * 2);
  __bf16* Xh   = (__bf16*)alloc(XY_ELEMS * 2);
  __bf16* Xl   = (__bf16*)alloc(XY_ELEMS * 2);
  __bf16* Qh   = (__bf16*)alloc(QKV_ELEMS * 2);
  __bf16* Ql   = (__bf16*)alloc(QKV_ELEMS * 2);
  __bf16* Kh   = (__bf16*)alloc(QKV_ELEMS * 2);
  __bf16* Kl   = (__bf16*)alloc(QKV_ELEMS * 2);
  __bf16* Vth  = (__bf16*)alloc(QKV_ELEMS * 2);
  __bf16* Vtl  = (__bf16*)alloc(QKV_ELEMS * 2);
  __bf16* Yh   = (__bf16*)alloc(XY_ELEMS * 2);
  __bf16* Yl   = (__bf16*)alloc(XY_ELEMS * 2);

  split_kernel<<<(unsigned)(XY_ELEMS / 256), 256, 0, stream>>>(
      x, Xh, Xl, (unsigned)XY_ELEMS);
  split_transpose_kernel<<<(3072 * 1024) / 256, 256, 0, stream>>>(
      W_attn, WAth, WAtl, 1024, 3072);
  split_transpose_kernel<<<(1024 * 1024) / 256, 256, 0, stream>>>(
      W_proj, WPth, WPtl, 1024, 1024);
  qkv_rope_kernel<<<dim3(32, 24), 256, 0, stream>>>(
      Xh, Xl, WAth, WAtl, b_attn, rope, Qh, Ql, Kh, Kl, Vth, Vtl);
  attn_kernel<<<dim3(8, 32), 256, 0, stream>>>(Qh, Ql, Kh, Kl, Vth, Vtl, Yh,
                                               Yl);
  proj_kernel<<<dim3(32, 8), 256, 0, stream>>>(Yh, Yl, WPth, WPtl, b_proj,
                                               (float*)d_out);
}
